// MultiHeadAttention_70111046140323
// MI455X (gfx1250) — matrix operand order repaired and checked by reading, not compiled, not run on hardware
//
#include <hip/hip_runtime.h>

#define DIM   1024
#define SEQ   2048
#define BATCH 2
#define HEADS 16
#define DK    64
#define MROWS (BATCH*SEQ)

typedef __attribute__((ext_vector_type(16))) __bf16 v16bf;
typedef __attribute__((ext_vector_type(8)))  __bf16 v8bf;
typedef __attribute__((ext_vector_type(8)))  float  v8f;
typedef unsigned int u32x4 __attribute__((ext_vector_type(4)));
typedef int          i32x4 __attribute__((ext_vector_type(4)));
typedef int          i32x8 __attribute__((ext_vector_type(8)));

#if defined(__has_builtin)
#if __has_builtin(__builtin_amdgcn_tensor_load_to_lds) && __has_builtin(__builtin_amdgcn_s_wait_tensorcnt)
#define USE_TDM 1
#endif
#endif
#ifndef USE_TDM
#define USE_TDM 0
#endif

__device__ __forceinline__ unsigned short f2bf(float f) {
  unsigned int u = __float_as_uint(f);
  u += 0x7FFFu + ((u >> 16) & 1u);   // round-to-nearest-even
  return (unsigned short)(u >> 16);
}

// ---------------------------------------------------------------- convert
__global__ void cvt_bf16(const float* __restrict__ src,
                         unsigned short* __restrict__ dst, int n) {
  int i = blockIdx.x * blockDim.x + threadIdx.x;
  int stride = gridDim.x * blockDim.x;
  for (; i < n; i += stride) dst[i] = f2bf(src[i]);
}

#if USE_TDM
// 2D TDM descriptor: tile_h rows x tile_w bf16 elements, row stride in elements.
// D# layout per CDNA5 ISA.
__device__ __forceinline__ void tdm_2d_load(unsigned lds_off, const void* gptr,
                                            int tile_w, int tile_h, int row_stride) {
  unsigned long long ga = (unsigned long long)(size_t)gptr;
  u32x4 g0;
  g0[0] = 1u;                                       // count=1, user mode
  g0[1] = lds_off;                                  // lds_addr [63:32]
  g0[2] = (unsigned)(ga & 0xFFFFFFFFull);           // global_addr low
  g0[3] = (unsigned)((ga >> 32) & 0x01FFFFFFull) | (2u << 30);  // addr hi | type=2
  const unsigned td = 0x7FFFFFFFu;                  // huge tensor dims: never OOB-clip
  unsigned long long s0 = (unsigned long long)row_stride;
  i32x8 g1;
  g1[0] = (int)(1u << 16);                          // workgroup_mask=0, data_size=1 (2B)
  g1[1] = (int)((td & 0xFFFFu) << 16);              // tensor_dim0 lo16 @48
  g1[2] = (int)((td >> 16) | ((td & 0xFFFFu) << 16));  // dim0 hi | dim1 lo
  g1[3] = (int)((td >> 16) | ((unsigned)tile_w << 16));// dim1 hi | tile_dim0
  g1[4] = (int)(unsigned)tile_h;                    // tile_dim1 | tile_dim2=0
  g1[5] = (int)(unsigned)(s0 & 0xFFFFFFFFull);      // dim0_stride lo32
  g1[6] = (int)(unsigned)((s0 >> 32) & 0xFFFFull);  // dim0_stride hi16 | dim1_stride lo=0
  g1[7] = 0;
  i32x4 g2 = {0, 0, 0, 0};
  i32x4 g3 = {0, 0, 0, 0};
#if __clang_major__ >= 23
  i32x8 g4 = {0, 0, 0, 0, 0, 0, 0, 0};
  __builtin_amdgcn_tensor_load_to_lds(g0, g1, g2, g3, g4, 0);
#else
  __builtin_amdgcn_tensor_load_to_lds(g0, g1, g2, g3, 0);
#endif
}
#endif

// ---------------------------------------------------------------- epilogue store
// mode 1: RoPE -> [b,h,s,dk] bf16; mode 2: plain -> [b,h,dk,s] bf16; mode 3: f32 [m][e]
__device__ __forceinline__ void store_tile(v8f acc, int m0, int n0, int mode,
                                           void* OutP, const int* __restrict__ tpos,
                                           int hl, int ln) {
  int e = n0 + ln;
  int head = e >> 6, je = e & 63;
  int b_ = m0 >> 11;                 // row tile never straddles batches
  int bh = b_ * HEADS + head;
  if (mode == 1) {
    unsigned short* Out = (unsigned short*)OutP;
    int p = je >> 1;
    float inv = __expf(-(float)p * (2.0f * 9.210340371976184f / 64.0f));
    #pragma unroll
    for (int j = 0; j < 8; ++j) {
      int m = m0 + j + 8 * hl;
      int s = m & (SEQ - 1);
      float ang = (float)tpos[s] * inv;
      float sn, cs; __sincosf(ang, &sn, &cs);
      float v = acc[j];
      float partner = __shfl_xor(v, 1, 32);
      float o = ((e & 1) == 0) ? (v * cs - partner * sn)
                               : (partner * sn + v * cs);
      Out[((size_t)bh * SEQ + s) * DK + je] = f2bf(o);
    }
  } else if (mode == 2) {
    unsigned short* Out = (unsigned short*)OutP;
    #pragma unroll
    for (int j = 0; j < 8; ++j) {
      int m = m0 + j + 8 * hl;
      int s = m & (SEQ - 1);
      Out[((size_t)bh * DK + je) * SEQ + s] = f2bf(acc[j]);
    }
  } else {
    float* Out = (float*)OutP;
    #pragma unroll
    for (int j = 0; j < 8; ++j)
      Out[(size_t)(m0 + j + 8 * hl) * DIM + n0 + ln] = acc[j];
  }
}

// ---------------------------------------------------------------- block-tiled GEMM
// C[m,e] = sum_d A[m,d]*W[e,d].  BM=128 BN=128 BK=32, 8 waves, TDM -> LDS,
// double buffered; each wave computes 32x64 (2x4 WMMA tiles).
__global__ void __launch_bounds__(256)
gemm_bf16(const unsigned short* __restrict__ Ag,   // [Mrows][DIM]
          const unsigned short* __restrict__ Wg,   // [DIM][DIM]
          void* __restrict__ OutP,
          const int* __restrict__ tpos, int mode) {
  __shared__ __align__(16) unsigned short As[2][128 * 32];
  __shared__ __align__(16) unsigned short Bs[2][128 * 32];
  int wid = threadIdx.x >> 5, lane = threadIdx.x & 31;
  int hl = lane >> 4, ln = lane & 15;
  const int nb = DIM / 128;                  // 8 N blocks
  int bidm = blockIdx.x / nb, bidn = blockIdx.x % nb;
  int gm0 = bidm * 128, gn0 = bidn * 128;
  int wm = wid >> 1, wn = wid & 1;
  int r0 = wm * 32, c0 = wn * 64;            // this wave's block-local origin

  v8f acc[2][4];
  #pragma unroll
  for (int s = 0; s < 2; ++s)
    #pragma unroll
    for (int n = 0; n < 4; ++n) { v8f z = {}; acc[s][n] = z; }

#if USE_TDM
  if (wid == 0) {
    tdm_2d_load((unsigned)(size_t)&As[0][0], Ag + (size_t)gm0 * DIM, 32, 128, DIM);
    tdm_2d_load((unsigned)(size_t)&Bs[0][0], Wg + (size_t)gn0 * DIM, 32, 128, DIM);
  }
  for (int kt = 0; kt < DIM / 32; ++kt) {
    int cur = kt & 1;
    if (wid == 0) __builtin_amdgcn_s_wait_tensorcnt(0);
    __syncthreads();                         // cur ready; prev compute drained
    if (wid == 0 && kt + 1 < DIM / 32) {     // prefetch next K-slab into other buffer
      int k0 = (kt + 1) * 32;
      tdm_2d_load((unsigned)(size_t)&As[cur ^ 1][0],
                  Ag + (size_t)gm0 * DIM + k0, 32, 128, DIM);
      tdm_2d_load((unsigned)(size_t)&Bs[cur ^ 1][0],
                  Wg + (size_t)gn0 * DIM + k0, 32, 128, DIM);
    }
    const unsigned short* Ab = &As[cur][0];
    const unsigned short* Bb = &Bs[cur][0];
    v16bf a[2], b[4];
    #pragma unroll
    for (int s = 0; s < 2; ++s) {
      const unsigned short* row = Ab + (size_t)(r0 + 16 * s + ln) * 32;
      ((v8bf*)&a[s])[0] = *(const v8bf*)(row + 16 * hl);
      ((v8bf*)&a[s])[1] = *(const v8bf*)(row + 8 + 16 * hl);
    }
    #pragma unroll
    for (int n = 0; n < 4; ++n)
      b[n] = *(const v16bf*)(Bb + (size_t)(c0 + 16 * n + ln) * 32 + 16 * hl);
    #pragma unroll
    for (int s = 0; s < 2; ++s)
      #pragma unroll
      for (int n = 0; n < 4; ++n)
        acc[s][n] = __builtin_amdgcn_wmma_f32_16x16x32_bf16(
            false, a[s], false, b[n], (short)0, acc[s][n], false, false);
  }
#else
  // fallback: per-wave streaming loads (no TDM builtin on this toolchain)
  const unsigned short* Arow0 = Ag + (size_t)(gm0 + r0 + ln) * DIM;
  const unsigned short* Arow1 = Ag + (size_t)(gm0 + r0 + 16 + ln) * DIM;
  for (int k0 = 0; k0 < DIM; k0 += 32) {
    v16bf a[2], b[4];
    ((v8bf*)&a[0])[0] = *(const v8bf*)(Arow0 + k0 + 16 * hl);
    ((v8bf*)&a[0])[1] = *(const v8bf*)(Arow0 + k0 + 8 + 16 * hl);
    ((v8bf*)&a[1])[0] = *(const v8bf*)(Arow1 + k0 + 16 * hl);
    ((v8bf*)&a[1])[1] = *(const v8bf*)(Arow1 + k0 + 8 + 16 * hl);
    #pragma unroll
    for (int n = 0; n < 4; ++n)
      b[n] = *(const v16bf*)(Wg + (size_t)(gn0 + c0 + 16 * n + ln) * DIM + k0 + 16 * hl);
    #pragma unroll
    for (int s = 0; s < 2; ++s)
      #pragma unroll
      for (int n = 0; n < 4; ++n)
        acc[s][n] = __builtin_amdgcn_wmma_f32_16x16x32_bf16(
            false, a[s], false, b[n], (short)0, acc[s][n], false, false);
  }
#endif

  #pragma unroll
  for (int s = 0; s < 2; ++s)
    #pragma unroll
    for (int n = 0; n < 4; ++n)
      store_tile(acc[s][n], gm0 + r0 + 16 * s, gn0 + c0 + 16 * n,
                 mode, OutP, tpos, hl, ln);
}

// ---------------------------------------------------------------- flash attention
__global__ void __launch_bounds__(256)
attn_kernel(const unsigned short* __restrict__ Qb,   // [b,h,s,dk]
            const unsigned short* __restrict__ Kb,   // [b,h,s,dk]
            const unsigned short* __restrict__ Vb,   // [b,h,dk,s]
            unsigned short* __restrict__ AOb) {      // [b,s,e] bf16
  __shared__ __align__(16) unsigned short pt[8][16 * 32];
  int wid  = threadIdx.x >> 5;
  int lane = threadIdx.x & 31;
  int hl = lane >> 4, ln = lane & 15;
  int gw = blockIdx.x * (blockDim.x >> 5) + wid;
  int qt = gw & 127;
  int bh = gw >> 7;
  int q0 = qt << 4;

  const unsigned short* Qh = Qb + (size_t)bh * SEQ * DK;
  const unsigned short* Kh = Kb + (size_t)bh * SEQ * DK;
  const unsigned short* Vh = Vb + (size_t)bh * DK * SEQ;

  v16bf qa[2];
  const unsigned short* qrow = Qh + (size_t)(q0 + ln) * DK;
  #pragma unroll
  for (int c = 0; c < 2; ++c) {
    ((v8bf*)&qa[c])[0] = *(const v8bf*)(qrow + 32 * c + 16 * hl);
    ((v8bf*)&qa[c])[1] = *(const v8bf*)(qrow + 32 * c + 8 + 16 * hl);
  }

  float mrow[8], lrow[8];
  v8f o[4];
  #pragma unroll
  for (int j = 0; j < 8; ++j) { mrow[j] = -INFINITY; lrow[j] = 0.0f; }
  #pragma unroll
  for (int t = 0; t < 4; ++t) { v8f z = {}; o[t] = z; }

  int kmax = q0 + 15;
  for (int kbase = 0; kbase <= kmax; kbase += 32) {
    v8f c0 = {}, c1 = {};
    const unsigned short* k0row = Kh + (size_t)(kbase + ln) * DK;
    const unsigned short* k1row = Kh + (size_t)(kbase + 16 + ln) * DK;
    #pragma unroll
    for (int ch = 0; ch < 2; ++ch) {
      v16bf kb0 = *(const v16bf*)(k0row + 32 * ch + 16 * hl);
      v16bf kb1 = *(const v16bf*)(k1row + 32 * ch + 16 * hl);
      c0 = __builtin_amdgcn_wmma_f32_16x16x32_bf16(false, qa[ch], false, kb0,
                                                   (short)0, c0, false, false);
      c1 = __builtin_amdgcn_wmma_f32_16x16x32_bf16(false, qa[ch], false, kb1,
                                                   (short)0, c1, false, false);
    }
    float p0[8], p1[8];
    #pragma unroll
    for (int j = 0; j < 8; ++j) {
      int q = q0 + j + 8 * hl;
      float s0 = (kbase + ln      <= q) ? c0[j] * 0.125f : -INFINITY;
      float s1 = (kbase + 16 + ln <= q) ? c1[j] * 0.125f : -INFINITY;
      float mx = fmaxf(s0, s1);
      #pragma unroll
      for (int d = 1; d < 16; d <<= 1) mx = fmaxf(mx, __shfl_xor(mx, d, 32));
      float mnew = fmaxf(mrow[j], mx);
      float corr = __expf(mrow[j] - mnew);
      float e0 = __expf(s0 - mnew);
      float e1 = __expf(s1 - mnew);
      float rs = e0 + e1;
      #pragma unroll
      for (int d = 1; d < 16; d <<= 1) rs += __shfl_xor(rs, d, 32);
      lrow[j] = lrow[j] * corr + rs;
      mrow[j] = mnew;
      o[0][j] *= corr; o[1][j] *= corr; o[2][j] *= corr; o[3][j] *= corr;
      p0[j] = e0; p1[j] = e1;
    }
    unsigned short* myp = pt[wid];
    #pragma unroll
    for (int j = 0; j < 8; ++j) {
      int m = j + 8 * hl;
      myp[m * 32 + ln]      = f2bf(p0[j]);
      myp[m * 32 + 16 + ln] = f2bf(p1[j]);
    }
    asm volatile("" ::: "memory");        // same-wave DS is executed in order
    v16bf pa;
    ((v8bf*)&pa)[0] = *(const v8bf*)(myp + ln * 32 + 16 * hl);
    ((v8bf*)&pa)[1] = *(const v8bf*)(myp + ln * 32 + 8 + 16 * hl);
    #pragma unroll
    for (int t = 0; t < 4; ++t) {
      v16bf vv = *(const v16bf*)(Vh + (size_t)(16 * t + ln) * SEQ + kbase + 16 * hl);
      o[t] = __builtin_amdgcn_wmma_f32_16x16x32_bf16(false, pa, false, vv,
                                                     (short)0, o[t], false, false);
    }
  }

  int b_ = bh >> 4, head = bh & 15;
  #pragma unroll
  for (int j = 0; j < 8; ++j) {
    float rinv = 1.0f / lrow[j];
    int s = q0 + j + 8 * hl;
    size_t rb = ((size_t)b_ * SEQ + s) * DIM + head * DK;
    #pragma unroll
    for (int t = 0; t < 4; ++t)
      AOb[rb + 16 * t + ln] = f2bf(o[t][j] * rinv);
  }
}

// ---------------------------------------------------------------- launch
extern "C" void kernel_launch(void* const* d_in, const int* in_sizes, int n_in,
                              void* d_out, int out_size, void* d_ws, size_t ws_size,
                              hipStream_t stream) {
  const float* x  = (const float*)d_in[0];
  const float* Wq = (const float*)d_in[1];
  const float* Wk = (const float*)d_in[2];
  const float* Wv = (const float*)d_in[3];
  const float* Wo = (const float*)d_in[4];
  const int* tpos = (const int*)d_in[6];
  float* out = (float*)d_out;

  char* ws = (char*)d_ws;
  const size_t XB  = (size_t)MROWS * DIM * 2;
  const size_t WBS = (size_t)DIM * DIM * 2;
  const size_t QKV = (size_t)BATCH * HEADS * SEQ * DK * 2;
  unsigned short* xb  = (unsigned short*)(ws);
  unsigned short* wqb = (unsigned short*)(ws + XB);
  unsigned short* wkb = (unsigned short*)(ws + XB + WBS);
  unsigned short* wvb = (unsigned short*)(ws + XB + 2 * WBS);
  unsigned short* wob = (unsigned short*)(ws + XB + 3 * WBS);
  unsigned short* qb  = (unsigned short*)(ws + XB + 4 * WBS);
  unsigned short* kb  = (unsigned short*)(ws + XB + 4 * WBS + QKV);
  unsigned short* vb  = (unsigned short*)(ws + XB + 4 * WBS + 2 * QKV);
  unsigned short* aob = (unsigned short*)(ws + XB + 4 * WBS + 3 * QKV);

  cvt_bf16<<<1024, 256, 0, stream>>>(x,  xb,  MROWS * DIM);
  cvt_bf16<<<512,  256, 0, stream>>>(Wq, wqb, DIM * DIM);
  cvt_bf16<<<512,  256, 0, stream>>>(Wk, wkb, DIM * DIM);
  cvt_bf16<<<512,  256, 0, stream>>>(Wv, wvb, DIM * DIM);
  cvt_bf16<<<512,  256, 0, stream>>>(Wo, wob, DIM * DIM);

  // (4096/128) x (1024/128) = 256 workgroups, 8 waves each
  gemm_bf16<<<256, 256, 0, stream>>>(xb, wqb, (void*)qb, tpos, 1);
  gemm_bf16<<<256, 256, 0, stream>>>(xb, wkb, (void*)kb, tpos, 1);
  gemm_bf16<<<256, 256, 0, stream>>>(xb, wvb, (void*)vb, tpos, 2);

  attn_kernel<<<512, 256, 0, stream>>>(qb, kb, vb, aob);

  gemm_bf16<<<256, 256, 0, stream>>>(aob, wob, (void*)out, tpos, 3);
}
